// LHmmLm_6957847019520
// MI455X (gfx1250) — matrix operand order repaired and checked by reading, not compiled, not run on hardware
//
#include <hip/hip_runtime.h>
#include <stdint.h>

#define H  256
#define C  512
#define VV 32000
#define NB 64
#define TT 512

typedef __attribute__((ext_vector_type(8)))  float  v8f;
typedef __attribute__((ext_vector_type(16))) __bf16 v16bf;

union BF16x16 { int4 q[2]; v16bf v; };

__device__ __forceinline__ unsigned short f2bf(float f) {
  unsigned u = __builtin_bit_cast(unsigned, f);
  unsigned r = u + 0x7FFFu + ((u >> 16) & 1u);
  return (unsigned short)(r >> 16);
}

// ---------------- workspace layout (bytes) ----------------
static constexpr size_t WS_START  = 0;                      // 512 f32
static constexpr size_t WS_LSE    = 4096;                   // 512 f32
static constexpr size_t WS_EVID   = 8192;                   // 64 f32
static constexpr size_t WS_PM     = 16384;                  // 128*512 f32
static constexpr size_t WS_PS     = WS_PM + 128*512*4;      // 128*512 f32
static constexpr size_t WS_PT     = 540672;                 // 512*512 bf16 (P^T)
static constexpr size_t WS_FEBF   = WS_PT + (size_t)C*C*2;        // 512*256 bf16
static constexpr size_t WS_TEMBBF = WS_FEBF + (size_t)C*H*2;      // 32000*256 bf16
static constexpr size_t WS_EMT    = WS_TEMBBF + (size_t)VV*H*2;   // 32000*512 f32 (emission^T logits)

// ---------------- shared helpers ----------------
__device__ void res2_block(float* x, float* h1, const float* rw, const float* rb, int t) {
  for (int i = 0; i < 2; ++i) {
    const float* w1 = rw + (size_t)(i*2 + 0)*H*H;
    const float* w2 = rw + (size_t)(i*2 + 1)*H*H;
    const float* b1 = rb + (i*2 + 0)*H;
    const float* b2 = rb + (i*2 + 1)*H;
    float a1 = b1[t];
    #pragma unroll 8
    for (int h = 0; h < H; ++h) a1 += x[h] * w1[h*H + t];
    a1 = fmaxf(a1, 0.f);
    h1[t] = a1;
    __syncthreads();
    float a2 = b2[t];
    #pragma unroll 8
    for (int h = 0; h < H; ++h) a2 += h1[h] * w2[h*H + t];
    x[t] += fmaxf(a2, 0.f);
    __syncthreads();
  }
}

// ---------------- 1) start distribution ----------------
__global__ void start_kernel(const float* __restrict__ start_emb,
                             const float* __restrict__ start_w,
                             const float* __restrict__ start_b,
                             const float* __restrict__ res_w,
                             const float* __restrict__ res_b,
                             const float* __restrict__ nse,
                             float* __restrict__ start_out) {
  __shared__ float x[H], h1[H], lg[C], red[256];
  int t = threadIdx.x;
  float acc = start_b[t];
  #pragma unroll 8
  for (int h = 0; h < H; ++h) acc += start_emb[h] * start_w[h*H + t];
  x[t] = acc;
  __syncthreads();
  res2_block(x, h1, res_w, res_b, t);
  // logits = x @ nse^T
  for (int c = t; c < C; c += 256) {
    float a = 0.f;
    #pragma unroll 8
    for (int h = 0; h < H; ++h) a += x[h] * nse[(size_t)c*H + h];
    lg[c] = a;
  }
  __syncthreads();
  float m = fmaxf(lg[t], lg[t + 256]);
  red[t] = m; __syncthreads();
  for (int s = 128; s > 0; s >>= 1) { if (t < s) red[t] = fmaxf(red[t], red[t+s]); __syncthreads(); }
  float M = red[0]; __syncthreads();
  red[t] = __expf(lg[t] - M) + __expf(lg[t+256] - M); __syncthreads();
  for (int s = 128; s > 0; s >>= 1) { if (t < s) red[t] += red[t+s]; __syncthreads(); }
  float L = logf(red[0]);
  start_out[t]       = lg[t]       - M - L;
  start_out[t + 256] = lg[t + 256] - M - L;
}

// ---------------- 2) transition row-softmax -> P^T (bf16) ----------------
__global__ void trans_kernel(const float* __restrict__ se_mat,
                             const float* __restrict__ nse,
                             unsigned short* __restrict__ Pt) {
  __shared__ float se[H], lg[C], red[256];
  int t = threadIdx.x, c = blockIdx.x;
  se[t] = se_mat[(size_t)c*H + t];
  __syncthreads();
  for (int c2 = t; c2 < C; c2 += 256) {
    float a = 0.f;
    #pragma unroll 8
    for (int h = 0; h < H; ++h) a += se[h] * nse[(size_t)c2*H + h];
    lg[c2] = a;
  }
  __syncthreads();
  float m = fmaxf(lg[t], lg[t + 256]);
  red[t] = m; __syncthreads();
  for (int s = 128; s > 0; s >>= 1) { if (t < s) red[t] = fmaxf(red[t], red[t+s]); __syncthreads(); }
  float M = red[0]; __syncthreads();
  red[t] = __expf(lg[t] - M) + __expf(lg[t+256] - M); __syncthreads();
  for (int s = 128; s > 0; s >>= 1) { if (t < s) red[t] += red[t+s]; __syncthreads(); }
  float inv = 1.f / red[0];
  // P^T[c', c] = softmax_c'(logits)[c']
  Pt[(size_t)t        *C + c] = f2bf(__expf(lg[t]       - M) * inv);
  Pt[(size_t)(t + 256)*C + c] = f2bf(__expf(lg[t + 256] - M) * inv);
}

// ---------------- 3) terminal MLP -> fe (bf16) ----------------
__global__ void term_mlp_kernel(const float* __restrict__ pre,
                                const float* __restrict__ res_w,
                                const float* __restrict__ res_b,
                                unsigned short* __restrict__ febf) {
  __shared__ float x[H], h1[H];
  int t = threadIdx.x, c = blockIdx.x;
  x[t] = pre[(size_t)c*H + t];
  __syncthreads();
  res2_block(x, h1, res_w, res_b, t);
  febf[(size_t)c*H + t] = f2bf(x[t]);
}

// ---------------- 4) terminal_emb f32 -> bf16 ----------------
__global__ void cvt_temb_kernel(const float4* __restrict__ src,
                                ushort4* __restrict__ dst, int n4) {
  int i = blockIdx.x * blockDim.x + threadIdx.x;
  if (i < n4) {
    float4 f = src[i];
    ushort4 o; o.x = f2bf(f.x); o.y = f2bf(f.y); o.z = f2bf(f.z); o.w = f2bf(f.w);
    dst[i] = o;
  }
}

// ---------------- 5) emission logits: emT[v,c] = fe[c,:] . temb[v,:] ----------------
__global__ void emis_kernel(const unsigned short* __restrict__ febf,
                            const unsigned short* __restrict__ tembbf,
                            float* __restrict__ emT) {
  __shared__ float tr[8][16*17];
  int lane = threadIdx.x & 31, w = threadIdx.x >> 5;
  int half = lane >> 4, mn = lane & 15;
  int c0 = (blockIdx.y * 8 + w) * 16;
  int v0 = blockIdx.x * 16;
  const unsigned short* arow = febf   + (size_t)(c0 + mn) * H;  // A: fe rows (M=c)
  const unsigned short* brow = tembbf + (size_t)(v0 + mn) * H;  // B: temb^T cols (N=v)
  v8f acc = {};
  #pragma unroll
  for (int k0 = 0; k0 < H; k0 += 32) {
    BF16x16 A, B;
    A.q[0] = *(const int4*)(arow + k0 + 8*half);
    A.q[1] = *(const int4*)(arow + k0 + 16 + 8*half);
    B.q[0] = *(const int4*)(brow + k0 + 8*half);
    B.q[1] = *(const int4*)(brow + k0 + 16 + 8*half);
    acc = __builtin_amdgcn_wmma_f32_16x16x32_bf16(false, A.v, false, B.v,
                                                  (short)0, acc, false, false);
  }
  float* slab = tr[w];
  #pragma unroll
  for (int j = 0; j < 8; ++j) slab[(j + 8*half)*17 + mn] = acc[j];  // [M=c][N=v]
  __syncthreads();
  #pragma unroll
  for (int s = 0; s < 8; ++s) {
    int vloc = 2*s + half, cloc = mn;
    emT[(size_t)(v0 + vloc)*C + c0 + cloc] = slab[cloc*17 + vloc];  // coalesced in c
  }
}

// ---------------- 6) column logsumexp over emT (2-phase) ----------------
__global__ void lse_part_kernel(const float* __restrict__ emT,
                                float* __restrict__ pm, float* __restrict__ ps) {
  int c = threadIdx.x, ch = blockIdx.x;     // 128 chunks * 250 rows
  int v0 = ch * 250;
  float m = -1e30f, s = 0.f;
  for (int v = v0; v < v0 + 250; ++v) {
    float x = emT[(size_t)v*C + c];
    if (x > m) { s = s * __expf(m - x) + 1.f; m = x; }
    else       { s += __expf(x - m); }
  }
  pm[ch*C + c] = m; ps[ch*C + c] = s;
}

__global__ void lse_comb_kernel(const float* __restrict__ pm,
                                const float* __restrict__ ps,
                                float* __restrict__ lse) {
  int c = threadIdx.x;
  float m = -1e30f, s = 0.f;
  for (int ch = 0; ch < 128; ++ch) {
    float mc = pm[ch*C + c], sc = ps[ch*C + c];
    if (mc > m) { s = s * __expf(m - mc) + sc; m = mc; }
    else        { s += sc * __expf(mc - m); }
  }
  lse[c] = m + logf(s);
}

// ---------------- 7) persistent HMM scan (4 WGs x 16 batch rows) ----------------
__global__ void scan_kernel(const int* __restrict__ text,
                            const float* __restrict__ start,
                            const unsigned short* __restrict__ Pt,
                            const float* __restrict__ emT,
                            const float* __restrict__ lse,
                            float* __restrict__ evid) {
  __shared__ float albuf[16*C];                       // alpha  [row][state]
  __shared__ __align__(16) unsigned short abf[16*C];  // exp(alpha-m) bf16 (A matrix)
  __shared__ float mrow[16];
  __shared__ int   tok[16];
  int lane = threadIdx.x & 31, r = threadIdx.x >> 5;  // wave r owns batch row r
  int half = lane >> 4, mn = lane & 15;
  int n = blockIdx.x * 16 + r;

  int x0 = text[n*TT];
  for (int c = lane; c < C; c += 32)
    albuf[r*C + c] = start[c] + emT[(size_t)x0*C + c] - lse[c];
  __syncthreads();

  for (int t = 1; t < TT; ++t) {
    // phase 1: per-row max + exp into bf16 A
    float mx = -1e30f;
    for (int c = lane; c < C; c += 32) mx = fmaxf(mx, albuf[r*C + c]);
    #pragma unroll
    for (int off = 16; off > 0; off >>= 1) mx = fmaxf(mx, __shfl_xor(mx, off, 32));
    for (int c = lane; c < C; c += 32) abf[r*C + c] = f2bf(__expf(albuf[r*C + c] - mx));
    if (lane == 0) { mrow[r] = mx; tok[r] = text[n*TT + t]; }
    __syncthreads();

    // phase 2: [16 x 512] = exp(alpha-m) @ P  via bf16 WMMA; wave r does 2 col tiles
    float nv[2][8];
    const unsigned short* arow = abf + mn*C;          // A row M = mn (batch row)
    #pragma unroll
    for (int tile = 0; tile < 2; ++tile) {
      int cp0 = (2*r + tile) * 16;
      const unsigned short* brow = Pt + (size_t)(cp0 + mn)*C;  // B col N = mn (c')
      v8f acc = {};
      #pragma unroll
      for (int k0 = 0; k0 < C; k0 += 32) {
        BF16x16 A, B;
        A.q[0] = *(const int4*)(arow + k0 + 8*half);
        A.q[1] = *(const int4*)(arow + k0 + 16 + 8*half);
        B.q[0] = *(const int4*)(brow + k0 + 8*half);
        B.q[1] = *(const int4*)(brow + k0 + 16 + 8*half);
        acc = __builtin_amdgcn_wmma_f32_16x16x32_bf16(false, A.v, false, B.v,
                                                      (short)0, acc, false, false);
      }
      #pragma unroll
      for (int j = 0; j < 8; ++j) {
        int M = j + 8*half;                           // batch row
        float e = emT[(size_t)tok[M]*C + cp0 + mn] - lse[cp0 + mn];
        nv[tile][j] = logf(fmaxf(acc[j], 1e-30f)) + mrow[M] + e;
      }
    }
    __syncthreads();   // all A reads done before alpha overwrite
    #pragma unroll
    for (int tile = 0; tile < 2; ++tile)
      #pragma unroll
      for (int j = 0; j < 8; ++j)
        albuf[(j + 8*half)*C + (2*r + tile)*16 + mn] = nv[tile][j];
    __syncthreads();
  }

  // final: per-row logsumexp -> evidence contribution
  float mx = -1e30f;
  for (int c = lane; c < C; c += 32) mx = fmaxf(mx, albuf[r*C + c]);
  #pragma unroll
  for (int off = 16; off > 0; off >>= 1) mx = fmaxf(mx, __shfl_xor(mx, off, 32));
  float s = 0.f;
  for (int c = lane; c < C; c += 32) s += __expf(albuf[r*C + c] - mx);
  #pragma unroll
  for (int off = 16; off > 0; off >>= 1) s += __shfl_xor(s, off, 32);
  if (lane == 0) evid[n] = mx + logf(s);
}

// ---------------- 8) final fixed-order sum ----------------
__global__ void final_kernel(const float* __restrict__ evid, float* __restrict__ out) {
  if (threadIdx.x == 0 && blockIdx.x == 0) {
    float s = 0.f;
    for (int i = 0; i < NB; ++i) s += evid[i];
    out[0] = s;
  }
}

extern "C" void kernel_launch(void* const* d_in, const int* in_sizes, int n_in,
                              void* d_out, int out_size, void* d_ws, size_t ws_size,
                              hipStream_t stream) {
  const int*   text     = (const int*)  d_in[0];
  const float* start_e  = (const float*)d_in[1];
  const float* start_w  = (const float*)d_in[2];
  const float* start_b  = (const float*)d_in[3];
  const float* start_rw = (const float*)d_in[4];
  const float* start_rb = (const float*)d_in[5];
  const float* state_e  = (const float*)d_in[6];
  const float* next_e   = (const float*)d_in[7];
  const float* pre_e    = (const float*)d_in[8];
  const float* term_rw  = (const float*)d_in[9];
  const float* term_rb  = (const float*)d_in[10];
  const float* term_e   = (const float*)d_in[11];

  char* ws = (char*)d_ws;
  float*          w_start = (float*)(ws + WS_START);
  float*          w_lse   = (float*)(ws + WS_LSE);
  float*          w_evid  = (float*)(ws + WS_EVID);
  float*          w_pm    = (float*)(ws + WS_PM);
  float*          w_ps    = (float*)(ws + WS_PS);
  unsigned short* w_pt    = (unsigned short*)(ws + WS_PT);
  unsigned short* w_febf  = (unsigned short*)(ws + WS_FEBF);
  unsigned short* w_tembb = (unsigned short*)(ws + WS_TEMBBF);
  float*          w_emt   = (float*)(ws + WS_EMT);

  start_kernel<<<1, 256, 0, stream>>>(start_e, start_w, start_b, start_rw, start_rb,
                                      next_e, w_start);
  trans_kernel<<<C, 256, 0, stream>>>(state_e, next_e, w_pt);
  term_mlp_kernel<<<C, 256, 0, stream>>>(pre_e, term_rw, term_rb, w_febf);
  cvt_temb_kernel<<<(VV*H/4 + 255)/256, 256, 0, stream>>>((const float4*)term_e,
                                                          (ushort4*)w_tembb, VV*H/4);
  emis_kernel<<<dim3(VV/16, 4), 256, 0, stream>>>(w_febf, w_tembb, w_emt);
  lse_part_kernel<<<128, C, 0, stream>>>(w_emt, w_pm, w_ps);
  lse_comb_kernel<<<1, C, 0, stream>>>(w_pm, w_ps, w_lse);
  scan_kernel<<<NB/16, 512, 0, stream>>>(text, w_start, w_pt, w_emt, w_lse, w_evid);
  final_kernel<<<1, 1, 0, stream>>>(w_evid, (float*)d_out);
}
